// SpectralConv1d_71760313582014
// MI455X (gfx1250) — matrix operand order repaired and checked by reading, not compiled, not run on hardware
//
#include <hip/hip_runtime.h>
#include <hip/hip_bf16.h>

// SpectralConv1d on MI455X (gfx1250).
// Roofline: weights = 1.07 GB read once -> ~46 us floor @ 23.3 TB/s; FFTs factored
// radix-256 so their GEMM stages (~51 GFLOP-bf16 each with hi/lo split) hide under
// the weight stream. Matrix math via v_wmma_f32_16x16x32_bf16 (f32 accumulate);
// the 1 GB weight stream in `contract` is moved by the Tensor Data Mover
// (tensor_load_to_lds + s_wait_tensorcnt), double-buffered in LDS.
//
// Workspace layout (needs 144 MB):
//   [0,64MB)    Yr   (fwd stage1 real)   -- reused as Tr for inverse
//   [64,128MB)  Yi                        -- reused as Ti
//   [128,132MB) XFr  (x_ft real, 2048x256)
//   [132,136MB) XFi
//   [136,140MB) Gr   (scaled out_ft real)
//   [140,144MB) Gi

typedef __attribute__((ext_vector_type(16))) __bf16 v16bf;
typedef __attribute__((ext_vector_type(8)))  float  v8f;
typedef __attribute__((ext_vector_type(4)))  unsigned int v4u;
typedef __attribute__((ext_vector_type(8)))  int    v8i;
typedef __attribute__((ext_vector_type(4)))  int    v4i;

#define TWO_PI 6.28318530717958647692f

#if defined(__has_builtin)
#if __has_builtin(__builtin_amdgcn_tensor_load_to_lds) && __has_builtin(__builtin_amdgcn_s_wait_tensorcnt)
#define HAVE_TDM 1
#endif
#endif
#ifndef HAVE_TDM
#define HAVE_TDM 0
#endif

__device__ __forceinline__ v8f wmma_bf16(const v16bf& a, const v16bf& b, v8f c) {
  // D = A(16x32 bf16) x B(32x16 bf16) + C(16x16 f32)
  return __builtin_amdgcn_wmma_f32_16x16x32_bf16(
      /*neg_a=*/false, a, /*neg_b=*/false, b,
      /*c_mod=*/(short)0, c, /*reuse_a=*/false, /*reuse_b=*/false);
}

// A-operand slot->K order used here (contiguous, same as B; not the ISA table order): lanes 0-15 hold K 0..15;
// lanes 16-31 hold K 16..31 (slot s -> VGPR s/2).
__device__ __forceinline__ int a_slot_k(int s, int lane) {
  return s + ((lane & 16) ? 16 : 0);
}

#if HAVE_TDM
// LDS byte offset of a shared-memory pointer (flat LDS addr keeps the offset
// in the low 32 bits).
__device__ __forceinline__ unsigned lds_off(const void* p) {
  return (unsigned)(unsigned long long)(uintptr_t)p;
}

// Issue one TDM DMA: 2-D tile of 16 (contiguous floats) x 32 rows, row stride
// 256 floats, from global `gsrc` into LDS at byte offset `ldsb`.
// D# per CDNA5 ISA: group0 = {count=1, lds_addr,
// global_addr, type=2}; group1 = {data_size=4B, tensor dims = all-ones (no
// clipping), tile_dim0=16, tile_dim1=32, tensor_dim0_stride=256}.
__device__ __forceinline__ void tdm_load_tile(const float* gsrc, unsigned ldsb) {
  unsigned long long ga = (unsigned long long)(uintptr_t)gsrc;
  unsigned w0 = 1u;                                            // count=1
  unsigned w1 = (unsigned)__builtin_amdgcn_readfirstlane((int)ldsb);
  unsigned w2 = (unsigned)__builtin_amdgcn_readfirstlane((int)(unsigned)(ga & 0xffffffffull));
  unsigned w3 = (unsigned)__builtin_amdgcn_readfirstlane(
                    (int)(((unsigned)(ga >> 32) & 0x01ffffffu) | 0x80000000u)); // type=2
  v4u g0 = { w0, w1, w2, w3 };
  v8i g1 = { (int)0x00020000,   // data_size = 2 (4 bytes)
             (int)0xFFFF0000,   // tensor_dim0[15:0] = 0xFFFF (huge)
             (int)0xFFFFFFFF,   // tensor_dim0[31:16]=0xFFFF, tensor_dim1[15:0]=0xFFFF
             (int)0x0010FFFF,   // tensor_dim1[31:16]=0xFFFF, tile_dim0 = 16
             (int)32,           // tile_dim1 = 32, tile_dim2 = 0
             (int)256,          // tensor_dim0_stride = 256 elements
             0, 0 };            // stride high bits / tensor_dim1_stride = 0
  v4i gz = { 0, 0, 0, 0 };
#if defined(__clang_major__) && (__clang_major__ >= 23)
  v8i gz8 = { 0, 0, 0, 0, 0, 0, 0, 0 };
  __builtin_amdgcn_tensor_load_to_lds(g0, g1, gz, gz, gz8, 0);
#else
  __builtin_amdgcn_tensor_load_to_lds(g0, g1, gz, gz, 0);
#endif
}
#endif  // HAVE_TDM

// ---------------------------------------------------------------------------
// Stage F1: Y[r, J] = sum_p x[p*65536 + J] * exp(-2*pi*i*r*p/256),
//           J = q*256 + c in [0,65536). WMMA GEMM, M=r tile 16, K=p 256, N=J.
// ---------------------------------------------------------------------------
__global__ __launch_bounds__(256) void fwd_stage1(const float* __restrict__ x,
                                                  float* __restrict__ Yr,
                                                  float* __restrict__ Yi) {
  __shared__ __bf16 tch[4096], tcl[4096], tsh[4096], tsl[4096]; // [16 r][256 p]
  const int tid   = threadIdx.x;
  const int rtile = blockIdx.x & 15;   // grid = 16 * 512 blocks
  const int jblk  = blockIdx.x >> 4;
  const int rbase = rtile * 16;

  for (int i = tid; i < 4096; i += 256) {
    int rl = i >> 8, p = i & 255;
    int prod = ((rbase + rl) * p) & 255;
    float ang = (float)prod * (TWO_PI / 256.0f);
    float sv, cv;
    __sincosf(ang, &sv, &cv);
    __bf16 h;
    h = (__bf16)cv; tch[i] = h; tcl[i] = (__bf16)(cv - (float)h);
    h = (__bf16)sv; tsh[i] = h; tsl[i] = (__bf16)(sv - (float)h);
  }
  __syncthreads();

  const int lane  = tid & 31;
  const int wave  = tid >> 5;
  const int jbase = jblk * 128 + wave * 16;
  const int n     = lane & 15;
  const int rl    = lane & 15;
  const int khB   = (lane & 16) ? 16 : 0;

  v8f accR = {};  // cos accumulation
  v8f accI = {};  // sin accumulation (negated on store: e^{-i t} = cos - i sin)
  for (int kb = 0; kb < 256; kb += 32) {
    v16bf bh, bl, ach, acl, ash, asl;
#pragma unroll
    for (int s = 0; s < 16; ++s) {
      int kB = kb + s + khB;                       // B slot->K: s + 16*(lane>=16)
      float v = x[(size_t)kB * 65536 + jbase + n];
      __bf16 h = (__bf16)v;
      bh[s] = h; bl[s] = (__bf16)(v - (float)h);
      int idx = rl * 256 + (kb + a_slot_k(s, lane));
      ach[s] = tch[idx]; acl[s] = tcl[idx];
      ash[s] = tsh[idx]; asl[s] = tsl[idx];
    }
    accR = wmma_bf16(ach, bh, accR);
    accR = wmma_bf16(ach, bl, accR);
    accR = wmma_bf16(acl, bh, accR);
    accI = wmma_bf16(ash, bh, accI);
    accI = wmma_bf16(ash, bl, accI);
    accI = wmma_bf16(asl, bh, accI);
  }
#pragma unroll
  for (int g = 0; g < 8; ++g) {                    // D: VGPR g = rows g / g+8
    int row = rbase + g + ((lane & 16) ? 8 : 0);
    size_t o = (size_t)row * 65536 + jbase + n;
    Yr[o] = accR[g];
    Yi[o] = -accI[g];
  }
}

// ---------------------------------------------------------------------------
// Stage F2: XF[256j + r, c] = sum_q e^{-2*pi*i*(256j+r)*q/65536} * Y[r, q*256+c]
// ---------------------------------------------------------------------------
__global__ __launch_bounds__(256) void fwd_stage2(const float* __restrict__ Yr,
                                                  const float* __restrict__ Yi,
                                                  float* __restrict__ XFr,
                                                  float* __restrict__ XFi) {
  __shared__ float tc[256][8], ts[256][8];
  const int r   = blockIdx.x;
  const int tid = threadIdx.x;
  {
    int q = tid;
#pragma unroll
    for (int j = 0; j < 8; ++j) {
      int m = 256 * j + r;
      int prod = (m * q) & 65535;
      float ang = (float)prod * (TWO_PI / 65536.0f);
      float sv, cv; __sincosf(ang, &sv, &cv);
      tc[q][j] = cv; ts[q][j] = sv;
    }
  }
  __syncthreads();
  const int ch = tid;
  float ar[8] = {}, ai[8] = {};
  for (int q = 0; q < 256; ++q) {
    float yr = Yr[(size_t)r * 65536 + q * 256 + ch];
    float yi = Yi[(size_t)r * 65536 + q * 256 + ch];
#pragma unroll
    for (int j = 0; j < 8; ++j) {
      float cv = tc[q][j], sv = ts[q][j];
      ar[j] += cv * yr + sv * yi;   // Re{e^{-it}(yr + i yi)}
      ai[j] += cv * yi - sv * yr;   // Im{...}
    }
  }
#pragma unroll
  for (int j = 0; j < 8; ++j) {
    XFr[(size_t)(256 * j + r) * 256 + ch] = ar[j];
    XFi[(size_t)(256 * j + r) * 256 + ch] = ai[j];
  }
}

// ---------------------------------------------------------------------------
// Contraction (the 1 GB weight stream): per mode m,
//   F[m,o] = sum_ci XF[m,ci] * (Wr + i Wi)[m,ci,o];  G = s_m * F
// s_0 = 1/N, s_m = 2/N (folds irfft Hermitian weighting + 1/N).
// Weights are DMA'd by the Tensor Data Mover into LDS (double buffered,
// s_wait_tensorcnt pipelined); WMMA A rows = {Xr_hi, Xr_lo, Xi_hi, Xi_lo};
// 4 WMMAs per K-chunk against Wr_hi/Wr_lo/Wi_hi/Wi_lo (3-term hi/lo split).
// Block = 128 threads (4 waves); each wave owns 4 column tiles of 16.
// ---------------------------------------------------------------------------
__global__ __launch_bounds__(128) void contract(const float* __restrict__ XFr,
                                                const float* __restrict__ XFi,
                                                const float* __restrict__ Wr,
                                                const float* __restrict__ Wi,
                                                float* __restrict__ Gr,
                                                float* __restrict__ Gi) {
  __shared__ float sxr[256], sxi[256];
  __shared__ float wbuf[4][2][2][32 * 16];  // [wave][buf][re/im][k*16+n] = 32 KB
  const int m   = blockIdx.x;
  const int tid = threadIdx.x;
  for (int i = tid; i < 256; i += 128) {
    sxr[i] = XFr[(size_t)m * 256 + i];
    sxi[i] = XFi[(size_t)m * 256 + i];
  }
  __syncthreads();

  const int lane = tid & 31;
  const int wave = tid >> 5;
  const int row  = lane & 15;
  const int khB  = (lane & 16) ? 16 : 0;
  const size_t wbase = (size_t)m * 65536;

  for (int t = 0; t < 4; ++t) {
    const int nb = (wave * 4 + t) * 16;
    v8f c1 = {}, c2 = {}, c3 = {}, c4 = {};

#if HAVE_TDM
    // Prologue: DMA chunk 0 into buffer 0.
    tdm_load_tile(Wr + wbase + nb, lds_off(&wbuf[wave][0][0][0]));
    tdm_load_tile(Wi + wbase + nb, lds_off(&wbuf[wave][0][1][0]));
#endif
    for (int kc = 0; kc < 8; ++kc) {
      const int kb = kc * 32;
#if HAVE_TDM
      if (kc < 7) {  // DMA next chunk into the other buffer, then wait for cur
        const int nx = (kc + 1) & 1;
        tdm_load_tile(Wr + wbase + (size_t)(kb + 32) * 256 + nb,
                      lds_off(&wbuf[wave][nx][0][0]));
        tdm_load_tile(Wi + wbase + (size_t)(kb + 32) * 256 + nb,
                      lds_off(&wbuf[wave][nx][1][0]));
        __builtin_amdgcn_s_wait_tensorcnt(2);  // in-order: cur chunk resident
      } else {
        __builtin_amdgcn_s_wait_tensorcnt(0);
      }
      const float* pr = &wbuf[wave][kc & 1][0][0];
      const float* pi = &wbuf[wave][kc & 1][1][0];
#endif
      v16bf af, wrh, wrl, wih, wil;
#pragma unroll
      for (int s = 0; s < 16; ++s) {
        int kA = kb + a_slot_k(s, lane);
        float av = 0.0f;
        if (row == 0)      av = sxr[kA];
        else if (row == 1) { float f = sxr[kA]; av = f - (float)(__bf16)f; }
        else if (row == 2) av = sxi[kA];
        else if (row == 3) { float f = sxi[kA]; av = f - (float)(__bf16)f; }
        af[s] = (__bf16)av;

        int kB = s + khB;  // K within chunk
#if HAVE_TDM
        float wr = pr[kB * 16 + (lane & 15)];
        float wi = pi[kB * 16 + (lane & 15)];
#else
        size_t wo = wbase + (size_t)(kb + kB) * 256 + nb + (lane & 15);
        float wr = Wr[wo];
        float wi = Wi[wo];
#endif
        __bf16 h;
        h = (__bf16)wr; wrh[s] = h; wrl[s] = (__bf16)(wr - (float)h);
        h = (__bf16)wi; wih[s] = h; wil[s] = (__bf16)(wi - (float)h);
      }
      c1 = wmma_bf16(af, wrh, c1);
      c2 = wmma_bf16(af, wrl, c2);
      c3 = wmma_bf16(af, wih, c3);
      c4 = wmma_bf16(af, wil, c4);
    }
    if (lane < 16) {  // rows 0..3 of D live in c[0..3] for lanes 0-15
      float xr_wr = c1[0] + c1[1] + c2[0];
      float xi_wi = c3[2] + c3[3] + c4[2];
      float xr_wi = c3[0] + c3[1] + c4[0];
      float xi_wr = c1[2] + c1[3] + c2[2];
      float sm = (m == 0) ? (1.0f / 65536.0f) : (2.0f / 65536.0f);
      Gr[(size_t)m * 256 + nb + lane] = sm * (xr_wr - xi_wi);
      Gi[(size_t)m * 256 + nb + lane] = sm * (xr_wi + xi_wr);
    }
  }
}

// ---------------------------------------------------------------------------
// Stage I1: T[r, b*256+o] = e^{+2*pi*i*r*b/65536} * sum_j G[256j+r, o] e^{+2*pi*i*j*b/256}
// ---------------------------------------------------------------------------
__global__ __launch_bounds__(256) void inv_stage1(const float* __restrict__ Gr,
                                                  const float* __restrict__ Gi,
                                                  float* __restrict__ Tr,
                                                  float* __restrict__ Ti) {
  __shared__ float tjc[256][8], tjs[256][8], trc[256], trs[256];
  const int r   = blockIdx.x;
  const int tid = threadIdx.x;
  {
    int b = tid;
#pragma unroll
    for (int j = 0; j < 8; ++j) {
      int prod = (j * b) & 255;
      float ang = (float)prod * (TWO_PI / 256.0f);
      float sv, cv; __sincosf(ang, &sv, &cv);
      tjc[b][j] = cv; tjs[b][j] = sv;
    }
    int prod = (r * b) & 65535;
    float ang = (float)prod * (TWO_PI / 65536.0f);
    float sv, cv; __sincosf(ang, &sv, &cv);
    trc[b] = cv; trs[b] = sv;
  }
  __syncthreads();
  const int o = tid;
  float gr[8], gi[8];
#pragma unroll
  for (int j = 0; j < 8; ++j) {
    gr[j] = Gr[(size_t)(256 * j + r) * 256 + o];
    gi[j] = Gi[(size_t)(256 * j + r) * 256 + o];
  }
  for (int b = 0; b < 256; ++b) {
    float Sr = 0.0f, Si = 0.0f;
#pragma unroll
    for (int j = 0; j < 8; ++j) {
      float cv = tjc[b][j], sv = tjs[b][j];
      Sr += gr[j] * cv - gi[j] * sv;
      Si += gr[j] * sv + gi[j] * cv;
    }
    float cv = trc[b], sv = trs[b];
    size_t off = (size_t)r * 65536 + b * 256 + o;
    Tr[off] = Sr * cv - Si * sv;
    Ti[off] = Sr * sv + Si * cv;
  }
}

// ---------------------------------------------------------------------------
// Stage I2: out[a*65536 + J] = sum_r cos(2*pi*r*a/256)*Tr[r,J] - sin(...)*Ti[r,J]
// (real part of e^{+i} recombination). Same WMMA GEMM structure as fwd_stage1.
// ---------------------------------------------------------------------------
__global__ __launch_bounds__(256) void inv_stage2(const float* __restrict__ Trp,
                                                  const float* __restrict__ Tip,
                                                  float* __restrict__ out) {
  __shared__ __bf16 tch[4096], tcl[4096], tsh[4096], tsl[4096]; // [16 a][256 r]
  const int tid   = threadIdx.x;
  const int atile = blockIdx.x & 15;
  const int jblk  = blockIdx.x >> 4;
  const int abase = atile * 16;

  for (int i = tid; i < 4096; i += 256) {
    int al = i >> 8, rr = i & 255;
    int prod = ((abase + al) * rr) & 255;
    float ang = (float)prod * (TWO_PI / 256.0f);
    float sv, cv; __sincosf(ang, &sv, &cv);
    __bf16 h;
    h = (__bf16)cv; tch[i] = h; tcl[i] = (__bf16)(cv - (float)h);
    h = (__bf16)sv; tsh[i] = h; tsl[i] = (__bf16)(sv - (float)h);
  }
  __syncthreads();

  const int lane  = tid & 31;
  const int wave  = tid >> 5;
  const int jbase = jblk * 128 + wave * 16;
  const int n     = lane & 15;
  const int al    = lane & 15;
  const int khB   = (lane & 16) ? 16 : 0;

  v8f accC = {}, accS = {};
  for (int kb = 0; kb < 256; kb += 32) {
    v16bf trh, trl, tih, til, ach, acl, ash, asl;
#pragma unroll
    for (int s = 0; s < 16; ++s) {
      int kB = kb + s + khB;
      float tr = Trp[(size_t)kB * 65536 + jbase + n];
      float tv = Tip[(size_t)kB * 65536 + jbase + n];
      __bf16 h;
      h = (__bf16)tr; trh[s] = h; trl[s] = (__bf16)(tr - (float)h);
      h = (__bf16)tv; tih[s] = h; til[s] = (__bf16)(tv - (float)h);
      int idx = al * 256 + (kb + a_slot_k(s, lane));
      ach[s] = tch[idx]; acl[s] = tcl[idx];
      ash[s] = tsh[idx]; asl[s] = tsl[idx];
    }
    accC = wmma_bf16(ach, trh, accC);
    accC = wmma_bf16(ach, trl, accC);
    accC = wmma_bf16(acl, trh, accC);
    accS = wmma_bf16(ash, tih, accS);
    accS = wmma_bf16(ash, til, accS);
    accS = wmma_bf16(asl, tih, accS);
  }
#pragma unroll
  for (int g = 0; g < 8; ++g) {
    int a = abase + g + ((lane & 16) ? 8 : 0);
    out[(size_t)a * 65536 + jbase + n] = accC[g] - accS[g];
  }
}

// ---------------------------------------------------------------------------
extern "C" void kernel_launch(void* const* d_in, const int* in_sizes, int n_in,
                              void* d_out, int out_size, void* d_ws, size_t ws_size,
                              hipStream_t stream) {
  (void)in_sizes; (void)n_in; (void)out_size; (void)ws_size;
  const float* x      = (const float*)d_in[0];
  const float* w_real = (const float*)d_in[1];
  const float* w_imag = (const float*)d_in[2];
  float* out = (float*)d_out;

  char* ws = (char*)d_ws;
  const size_t MB = 1024 * 1024;
  float* Yr  = (float*)(ws);                // 64 MB
  float* Yi  = (float*)(ws + 64 * MB);      // 64 MB
  float* XFr = (float*)(ws + 128 * MB);     // 4 MB
  float* XFi = (float*)(ws + 132 * MB);     // 4 MB
  float* Gr  = (float*)(ws + 136 * MB);     // 4 MB
  float* Gi  = (float*)(ws + 140 * MB);     // 4 MB
  float* Trp = Yr;                          // Y dead after fwd_stage2 -> reuse
  float* Tip = Yi;

  fwd_stage1<<<8192, 256, 0, stream>>>(x, Yr, Yi);
  fwd_stage2<<<256,  256, 0, stream>>>(Yr, Yi, XFr, XFi);
  contract  <<<2048, 128, 0, stream>>>(XFr, XFi, w_real, w_imag, Gr, Gi);
  inv_stage1<<<256,  256, 0, stream>>>(Gr, Gi, Trp, Tip);
  inv_stage2<<<8192, 256, 0, stream>>>(Trp, Tip, out);
}
